// SineModelWrapper_86612310491556
// MI455X (gfx1250) — matrix operand order repaired and checked by reading, not compiled, not run on hardware
//
#include <hip/hip_runtime.h>

typedef __attribute__((ext_vector_type(16))) _Float16 v16h;
typedef __attribute__((ext_vector_type(8)))  _Float16 v8h;
typedef __attribute__((ext_vector_type(8)))  float    v8f;

#define Bsz  256
#define Tsz  1024
#define Hsz  512
#define G4H  2048   // 4*H
#define Kcat 1024   // H (layer-0 h) + H (layer-1 h) concatenated per row
#define NWG  64
#define TPB  256
#define KB0  (Hsz / 32)    // 16 k-blocks, layer 0
#define KB1  (Kcat / 32)   // 32 k-blocks, layer 1
#define P0H  (4 * KB0 * 512)   // L0 LDS panel halves  (32768 = 64KB)
#define P1H  (4 * KB1 * 512)   // L1 LDS panel halves  (65536 = 128KB)
#define LDSB ((P0H + P1H) * 2) // 192KB dynamic LDS (needs CDNA5's 320KB WGP LDS)

__device__ __forceinline__ float sigm(float x)  { return 1.0f / (1.0f + __expf(-x)); }
__device__ __forceinline__ float tanha(float x) { return 1.0f - 2.0f / (__expf(2.0f * x) + 1.0f); }

// Build a 16-half fragment from two 16-byte loads (global_load_b128 / ds_load_b128).
__device__ __forceinline__ v16h ldfrag(const _Float16* p0, const _Float16* p1) {
  v8h lo = *(const v8h*)p0;
  v8h hi = *(const v8h*)p1;
  return __builtin_shufflevector(lo, hi, 0,1,2,3,4,5,6,7,8,9,10,11,12,13,14,15);
}

// Grid-wide barrier: monotonic counter, release on arrive / acquire on spin.
__device__ __forceinline__ void grid_sync(unsigned* cnt) {
  __syncthreads();
  if (threadIdx.x == 0) {
    __threadfence();
    unsigned prev = __hip_atomic_fetch_add(cnt, 1u, __ATOMIC_ACQ_REL, __HIP_MEMORY_SCOPE_AGENT);
    unsigned target = (prev / NWG + 1u) * NWG;
    while (__hip_atomic_load(cnt, __ATOMIC_ACQUIRE, __HIP_MEMORY_SCOPE_AGENT) < target)
      __builtin_amdgcn_s_sleep(1);
  }
  __syncthreads();
}

// 4-gate GEMM for one 16x16 block: A from global (h rows), B from LDS panel.
// LDS panel fragment layout, offsets in halves:
//   frag(g,kb) at (g*KB+kb)*512 ; lane reads 8 halves at lane*8 and at lane*8+256
//   (two sequential-per-lane 16B ds_load_b128s -> minimal bank conflicts).
template<int KB>
__device__ __forceinline__ void gemm_gates(const _Float16* __restrict__ A,
                                           const _Float16* __restrict__ Bl,
                                           int lane,
                                           v8f& ci, v8f& cf, v8f& cg, v8f& co) {
#pragma unroll 4
  for (int kb = 0; kb < KB; ++kb) {
    const int ko = kb * 32;
    v16h a = ldfrag(A + ko, A + ko + 16);
    const int o0 = (0 * KB + kb) * 512 + lane * 8;
    const int o1 = (1 * KB + kb) * 512 + lane * 8;
    const int o2 = (2 * KB + kb) * 512 + lane * 8;
    const int o3 = (3 * KB + kb) * 512 + lane * 8;
    v16h fI = ldfrag(Bl + o0, Bl + o0 + 256);
    v16h fF = ldfrag(Bl + o1, Bl + o1 + 256);
    v16h fG = ldfrag(Bl + o2, Bl + o2 + 256);
    v16h fO = ldfrag(Bl + o3, Bl + o3 + 256);
    ci = __builtin_amdgcn_wmma_f32_16x16x32_f16(false, a, false, fI, (short)0, ci, false, false);
    cf = __builtin_amdgcn_wmma_f32_16x16x32_f16(false, a, false, fF, (short)0, cf, false, false);
    cg = __builtin_amdgcn_wmma_f32_16x16x32_f16(false, a, false, fG, (short)0, cg, false, false);
    co = __builtin_amdgcn_wmma_f32_16x16x32_f16(false, a, false, fO, (short)0, co, false, false);
  }
}

// ---------------------------------------------------------------------------
// Persistent LSTM kernel. 64 WGs x 8 waves; WG w owns hidden block nb = w&31
// and batch blocks mb = (w>>5)*8 + wave -> all 8 waves share one weight panel,
// which is converted f32->f16 once into 192KB of LDS and stays resident for
// the whole recurrence (time-invariant weights).
// Phase k (k = -1..T-1), double-buffered h (buf p -> buf p^1):
//   L0 task (k < T-1): h0(k+1) = lstm0(h0(k), x[k+1])     [K=512 GEMM]
//   L1 task (k >= 0):  h1(k)   = lstm1([h0(k);h1(k-1)])   [K=1024 fused GEMM]
//                      out[:,k] += h1(k) @ Wf  (atomic, bf pre-seeded)
// One grid barrier per phase. Cell states c0/c1 live in registers.
// ---------------------------------------------------------------------------
__global__ __launch_bounds__(TPB)
void lstm_persistent(const float* __restrict__ input,   // [B,T]
                     const float* __restrict__ Wx0,     // [1,4H]
                     const float* __restrict__ Wh0,     // [H,4H]
                     const float* __restrict__ b0,      // [4H]
                     const float* __restrict__ Wx1,     // [H,4H]
                     const float* __restrict__ Wh1,     // [H,4H]
                     const float* __restrict__ b1,      // [4H]
                     const float* __restrict__ Wf,      // [H,1]
                     _Float16* __restrict__ hbuf,       // [2][B][Kcat] f16
                     float* __restrict__ out,           // [B,T]
                     unsigned* __restrict__ bar) {
  extern __shared__ _Float16 smem[];
  _Float16* B0l = smem;        // L0 panel: 4 gates x KB0 frags
  _Float16* B1l = smem + P0H;  // L1 panel: 4 gates x KB1 frags

  const int lane = threadIdx.x & 31;
  const int wave = threadIdx.x >> 5;
  const int nb   = blockIdx.x & 31;                 // hidden block (shared by WG)
  const int mb   = (blockIdx.x >> 5) * 8 + wave;    // batch block (per wave)
  const int col  = lane & 15;
  const int hs   = lane >> 4;                       // lane-half select
  const int n    = nb * 16 + col;                   // hidden index 0..511
  const int arow = mb * 16 + col;                   // A row this lane loads
  const int r0   = mb * 16 + hs * 8;                // first C row this lane owns

  // ---- Stage time-invariant weight panels into LDS (f32 -> f16 + transpose).
  // Linear index i == LDS fragment offset: i = ((g*KB+kb)*512) + part*256 + l16*8 + e
  for (int i = threadIdx.x; i < P0H; i += TPB) {
    const int e    = i & 7;
    const int l16  = (i >> 3) & 31;
    const int part = (i >> 8) & 1;
    const int fb   = i >> 9;             // g*KB0 + kb
    const int kb   = fb & (KB0 - 1);
    const int g    = fb / KB0;
    const int k    = kb * 32 + part * 16 + (l16 >> 4) * 8 + e;   // 0..511
    B0l[i] = (_Float16)Wh0[(size_t)k * G4H + g * Hsz + nb * 16 + (l16 & 15)];
  }
  for (int i = threadIdx.x; i < P1H; i += TPB) {
    const int e    = i & 7;
    const int l16  = (i >> 3) & 31;
    const int part = (i >> 8) & 1;
    const int fb   = i >> 9;             // g*KB1 + kb
    const int kb   = fb & (KB1 - 1);
    const int g    = fb / KB1;
    const int k    = kb * 32 + part * 16 + (l16 >> 4) * 8 + e;   // 0..1023
    const int cn   = g * Hsz + nb * 16 + (l16 & 15);
    const float v  = (k < Hsz) ? Wx1[(size_t)k * G4H + cn]
                               : Wh1[(size_t)(k - Hsz) * G4H + cn];
    B1l[i] = (_Float16)v;
  }
  __syncthreads();

  // Per-lane constants (one hidden column n).
  const float b0i = b0[n], b0f = b0[Hsz + n], b0g = b0[2 * Hsz + n], b0o = b0[3 * Hsz + n];
  const float b1i = b1[n], b1f = b1[Hsz + n], b1g = b1[2 * Hsz + n], b1o = b1[3 * Hsz + n];
  const float w0i = Wx0[n], w0f = Wx0[Hsz + n], w0g = Wx0[2 * Hsz + n], w0o = Wx0[3 * Hsz + n];
  const float wfv = Wf[n];

  float c0s[8], c1s[8];
#pragma unroll
  for (int v = 0; v < 8; ++v) { c0s[v] = 0.0f; c1s[v] = 0.0f; }

  int p = 0;
  for (int k = -1; k < Tsz; ++k) {
    const _Float16* hcur = hbuf + (size_t)p * Bsz * Kcat;
    _Float16*       hnxt = hbuf + (size_t)(p ^ 1) * Bsz * Kcat;
    const _Float16* Abase = hcur + (size_t)arow * Kcat + hs * 8;

    if (k < Tsz - 1) {  // ---- layer 0: produce h0(k+1)
      v8f ci = {}, cf = {}, cg = {}, co = {};
      gemm_gates<KB0>(Abase, B0l, lane, ci, cf, cg, co);
      const int t = k + 1;
#pragma unroll
      for (int v = 0; v < 8; ++v) {
        const float xv = input[(size_t)(r0 + v) * Tsz + t];
        const float gi = sigm (ci[v] + b0i + xv * w0i);
        const float gf = sigm (cf[v] + b0f + xv * w0f);
        const float gz = tanha(cg[v] + b0g + xv * w0g);
        const float go = sigm (co[v] + b0o + xv * w0o);
        const float cn = gf * c0s[v] + gi * gz;
        c0s[v] = cn;
        hnxt[(size_t)(r0 + v) * Kcat + n] = (_Float16)(go * tanha(cn));
      }
    }

    if (k >= 0) {  // ---- layer 1: produce h1(k) + fused projection
      v8f ci = {}, cf = {}, cg = {}, co = {};
      gemm_gates<KB1>(Abase, B1l, lane, ci, cf, cg, co);
      float pr[8];
#pragma unroll
      for (int v = 0; v < 8; ++v) {
        const float gi = sigm (ci[v] + b1i);
        const float gf = sigm (cf[v] + b1f);
        const float gz = tanha(cg[v] + b1g);
        const float go = sigm (co[v] + b1o);
        const float cn = gf * c1s[v] + gi * gz;
        c1s[v] = cn;
        const float hn = go * tanha(cn);
        hnxt[(size_t)(r0 + v) * Kcat + Hsz + n] = (_Float16)hn;
        pr[v] = hn * wfv;
      }
#pragma unroll
      for (int v = 0; v < 8; ++v) {  // reduce over 16 columns (lanes 0-15 / 16-31)
        float s = pr[v];
#pragma unroll
        for (int off = 1; off < 16; off <<= 1) s += __shfl_xor(s, off, 16);
        pr[v] = s;
      }
      if (col == 0) {
#pragma unroll
        for (int v = 0; v < 8; ++v)
          atomicAdd(&out[(size_t)(r0 + v) * Tsz + k], pr[v]);
      }
    }

    grid_sync(bar);
    p ^= 1;
  }
}

// ---------------------------------------------------------------------------
// Setup kernel (runs every launch -> deterministic state under graph replay).
// ---------------------------------------------------------------------------
__global__ void init_state(unsigned* __restrict__ hb_u32, float* __restrict__ out,
                           const float* __restrict__ bf, unsigned* __restrict__ bar) {
  const int idx = blockIdx.x * blockDim.x + threadIdx.x;
  if (idx < (2 * Bsz * Kcat) / 2) hb_u32[idx] = 0u;  // zero both h buffers (f16 pairs)
  if (idx < Bsz * Tsz) out[idx] = bf[0];             // pre-seed output with bias
  if (idx == 0) *bar = 0u;
}

// ---------------------------------------------------------------------------
extern "C" void kernel_launch(void* const* d_in, const int* in_sizes, int n_in,
                              void* d_out, int out_size, void* d_ws, size_t ws_size,
                              hipStream_t stream) {
  const float* input = (const float*)d_in[0];  // [B,T]
  const float* Wx0   = (const float*)d_in[1];  // [1,4H]
  const float* Wh0   = (const float*)d_in[2];  // [H,4H]
  const float* b0    = (const float*)d_in[3];  // [4H]
  const float* Wx1   = (const float*)d_in[4];  // [H,4H]
  const float* Wh1   = (const float*)d_in[5];  // [H,4H]
  const float* b1    = (const float*)d_in[6];  // [4H]
  const float* Wf    = (const float*)d_in[7];  // [H,1]
  const float* bf    = (const float*)d_in[8];  // [1]
  float* out = (float*)d_out;                  // [B,T]

  // Workspace layout (~1.05 MB): [0] barrier counter, [1KB] hbuf (1MB).
  char* ws = (char*)d_ws;
  unsigned* bar  = (unsigned*)ws;
  _Float16* hbuf = (_Float16*)(ws + 1024);

  init_state<<<(Bsz * Tsz + TPB - 1) / TPB, TPB, 0, stream>>>((unsigned*)hbuf, out, bf, bar);
  lstm_persistent<<<NWG, TPB, LDSB, stream>>>(input, Wx0, Wh0, b0, Wx1, Wh1, b1, Wf,
                                              hbuf, out, bar);

  (void)in_sizes; (void)n_in; (void)out_size; (void)ws_size;
}
